// NeuronLlamaCrossAttention_45956150067871
// MI455X (gfx1250) — matrix operand order repaired and checked by reading, not compiled, not run on hardware
//
#include <hip/hip_runtime.h>
#include <hip/hip_bf16.h>
#include <stdint.h>

typedef __attribute__((ext_vector_type(16))) __bf16 v16bf;
typedef __attribute__((ext_vector_type(4)))  __bf16 v4bf;
typedef __attribute__((ext_vector_type(8)))  float  v8f;

#define HIDDEN 4096
#define NHEADS 32
#define NKV    8
#define HD     128
#define SLEN   1024
#define NTOK   1601
#define SV     6404   // 4*1601

// LDS row strides: multiples of 8 elements (16B) so merged ds_load_b128 are
// aligned; chosen so 16 consecutive rows map to distinct bank groups.
#define STRA   72     // 144B = 9*16 ; bank = 36n mod 64 -> 16 distinct
#define STRP   56     // 112B = 7*16 ; bank = 28n mod 64 -> 16 distinct
#define STRK   136    // 272B = 17*16; bank = 4t  mod 64 -> 16 distinct

__device__ __forceinline__ float grp16_max(float v) {
    v = fmaxf(v, __shfl_xor(v, 1, 32));
    v = fmaxf(v, __shfl_xor(v, 2, 32));
    v = fmaxf(v, __shfl_xor(v, 4, 32));
    v = fmaxf(v, __shfl_xor(v, 8, 32));
    return v;
}
__device__ __forceinline__ float grp16_sum(float v) {
    v += __shfl_xor(v, 1, 32);
    v += __shfl_xor(v, 2, 32);
    v += __shfl_xor(v, 4, 32);
    v += __shfl_xor(v, 8, 32);
    return v;
}

// Async global->LDS copy of 8 bytes per lane (GLOBAL_LOAD_ASYNC_TO_LDS_B64,
// tracked by ASYNCcnt). ldsp: generic pointer into __shared__ (low 32 bits are
// the LDS byte address); gp: 64-bit global address.
__device__ __forceinline__ void async_copy_b64(const void* ldsp, const void* gp) {
    unsigned lds = (unsigned)(size_t)ldsp;
    unsigned long long ga = (unsigned long long)(size_t)gp;
    asm volatile("global_load_async_to_lds_b64 %0, %1, off"
                 :: "v"(lds), "v"(ga) : "memory");
}
__device__ __forceinline__ void async_wait0() {
    asm volatile("s_wait_asynccnt 0" ::: "memory");
}

// ---------------------------------------------------------------- convert
__global__ void cvt4_kernel(const float4* __restrict__ src, v4bf* __restrict__ dst, long n4) {
    long i = (long)blockIdx.x * blockDim.x + threadIdx.x;
    long stride = (long)gridDim.x * blockDim.x;
    for (; i < n4; i += stride) {
        float4 f = src[i];
        v4bf b;
        b[0] = (__bf16)f.x; b[1] = (__bf16)f.y; b[2] = (__bf16)f.z; b[3] = (__bf16)f.w;
        dst[i] = b;
    }
}

// ---------------------------------------------------------------- GEMM
// C(MxN) = A(MxK, bf16 row-major) @ B(KxN, bf16 row-major)
// block tile 128x128, BK=32, 256 threads = 8 waves (4 mw x 2 nw), wave tile 32x64.
// Rows >= Mvalid: source row is CLAMPED (WMMA rows are independent and those
// accumulator rows are never stored), so one uniform staging path.
// epi: 0 = plain, 1 = rmsnorm over 128-wide head slice (block N range == one head).
// headMajor: index outputs as [n/128][m][n%128] with head stride Mvalid*128.
__global__ __launch_bounds__(256)
void gemm_bf16_kernel(const __bf16* __restrict__ A, const __bf16* __restrict__ B,
                      float* __restrict__ f32C, __bf16* __restrict__ bfC,
                      const float* __restrict__ normw,
                      int N, int K, int Mvalid, int epi, int headMajor) {
    __shared__ __bf16 Al[128][STRA];
    __shared__ __bf16 Bt[128][STRA];   // transposed: [n][k]
    __shared__ float  red[128][2];

    const int tid  = threadIdx.x;
    const int lane = tid & 31;
    const int w    = tid >> 5;
    const int mw   = w >> 1, nw = w & 1;
    const int half = lane >> 4, l16 = lane & 15;

    const int rowBlock = blockIdx.x * 128;
    const int colBlock = blockIdx.y * 128;

    // staging coordinates
    const int ar0 = tid >> 3;             // 0..31
    const int ac0 = (tid & 7) * 4;        // 0..28
    const int bk  = (tid * 4) >> 7;       // 0..7   (first of 4 k-slices)
    const int bn0 = (tid * 4) & 127;

    // clamped source rows for the A tile (row-independent WMMA)
    int arow[4];
    for (int it = 0; it < 4; ++it) {
        int grow = rowBlock + ar0 + it * 32;
        arow[it] = grow < Mvalid ? grow : (Mvalid - 1);
    }

    v8f acc[2][4] = {};

    const int nkt = K >> 5;
    for (int kt = 0; kt < nkt; ++kt) {
        const __bf16* Ak = A + (size_t)kt * 32;
        const __bf16* Bk = B + (size_t)(kt * 32) * N + colBlock;
        // stage A tile 128x32: async global->LDS (8B per lane per iter)
        for (int it = 0; it < 4; ++it)
            async_copy_b64(&Al[ar0 + it * 32][ac0],
                           Ak + (size_t)arow[it] * K + ac0);
        // stage B tile 32x128 transposed into Bt[n][k]
        for (int it = 0; it < 4; ++it) {
            int k = bk + it * 8;
            v4bf val = *(const v4bf*)(Bk + (size_t)k * N + bn0);
            Bt[bn0 + 0][k] = val[0];
            Bt[bn0 + 1][k] = val[1];
            Bt[bn0 + 2][k] = val[2];
            Bt[bn0 + 3][k] = val[3];
        }
        // prefetch next k-tile (global_prefetch_b8)
        if (kt + 1 < nkt) {
            __builtin_prefetch(Ak + 32 + (size_t)arow[0] * K + ac0, 0, 1);
            __builtin_prefetch(Bk + (size_t)(32 + bk) * N + bn0, 0, 1);
        }
        async_wait0();
        __syncthreads();

        union Fr { v16bf v; v4bf c[4]; };
        Fr af[2], bfr[4];
        const int kb = half * 8;        // A-matrix K base per ISA layout
        for (int mi = 0; mi < 2; ++mi) {
            int r = mw * 32 + mi * 16 + l16;
            af[mi].c[0] = *(const v4bf*)&Al[r][kb];
            af[mi].c[1] = *(const v4bf*)&Al[r][kb + 4];
            af[mi].c[2] = *(const v4bf*)&Al[r][kb + 16];
            af[mi].c[3] = *(const v4bf*)&Al[r][kb + 20];
        }
        const int k0 = half * 8;        // B-matrix K base per ISA layout
        for (int ni = 0; ni < 4; ++ni) {
            int n = nw * 64 + ni * 16 + l16;
            bfr[ni].c[0] = *(const v4bf*)&Bt[n][k0];
            bfr[ni].c[1] = *(const v4bf*)&Bt[n][k0 + 4];
            bfr[ni].c[2] = *(const v4bf*)&Bt[n][k0 + 16];
            bfr[ni].c[3] = *(const v4bf*)&Bt[n][k0 + 20];
        }
        for (int mi = 0; mi < 2; ++mi)
            for (int ni = 0; ni < 4; ++ni)
                acc[mi][ni] = __builtin_amdgcn_wmma_f32_16x16x32_bf16(
                    false, af[mi].v, false, bfr[ni].v, (short)0, acc[mi][ni], false, false);
        __syncthreads();
    }

    // ---- epilogue
    float rms[2][8];
    if (epi == 1) {
        for (int mi = 0; mi < 2; ++mi)
            for (int r = 0; r < 8; ++r) {
                float p = 0.f;
                for (int ni = 0; ni < 4; ++ni) { float v = acc[mi][ni][r]; p += v * v; }
                p = grp16_sum(p);
                if (l16 == 0) red[mw * 32 + mi * 16 + half * 8 + r][nw] = p;
            }
        __syncthreads();
        for (int mi = 0; mi < 2; ++mi)
            for (int r = 0; r < 8; ++r) {
                int rl = mw * 32 + mi * 16 + half * 8 + r;
                float ss = red[rl][0] + red[rl][1];
                rms[mi][r] = rsqrtf(ss * (1.0f / 128.0f) + 1e-5f);
            }
    }
    const size_t headStride = (size_t)Mvalid * 128;
    for (int mi = 0; mi < 2; ++mi)
        for (int ni = 0; ni < 4; ++ni)
            for (int r = 0; r < 8; ++r) {
                int m = rowBlock + mw * 32 + mi * 16 + half * 8 + r;
                if (m >= Mvalid) continue;
                int n = colBlock + nw * 64 + ni * 16 + l16;
                float v = acc[mi][ni][r];
                if (epi == 1) v *= rms[mi][r] * normw[n & 127];
                size_t idx = headMajor ? (size_t)(n >> 7) * headStride + (size_t)m * 128 + (n & 127)
                                       : (size_t)m * N + n;
                if (f32C) f32C[idx] = v;
                if (bfC)  bfC[idx]  = (__bf16)v;
            }
}

// ---------------------------------------------------------------- flash attention
// block = (head h, q-tile of 128 rows), 8 waves x 16 rows each.
__global__ __launch_bounds__(256)
void xattn_kernel(const __bf16* __restrict__ Qb,   // [1024][4096] (head-concat cols)
                  const __bf16* __restrict__ Kb,   // [8][SV][128]
                  const __bf16* __restrict__ Vb,   // [8][SV][128]
                  __bf16* __restrict__ Ctx,        // [1024][4096]
                  const int* __restrict__ startp,
                  const int* __restrict__ nchp,
                  const float* __restrict__ hasimgp) {
    __shared__ __bf16 Kl[32][STRK];       // [t][d]
    __shared__ __bf16 Vt[128][STRA];      // [d][t] (transposed)
    __shared__ __bf16 Pl[8][16][STRP];    // per-wave P scratch

    const int tid = threadIdx.x, lane = tid & 31, w = tid >> 5;
    const int half = lane >> 4, l16 = lane & 15;
    const int h = blockIdx.x;
    const int kvh = h >> 2;
    const int s0 = blockIdx.y * 128 + w * 16;

    const int start = startp[0];
    int nch = nchp[0]; if (nch < 1) nch = 1; if (nch > 4) nch = 4;
    const float hasimg = hasimgp[0];
    const int Lv = nch * NTOK;

    union Fr { v16bf v; v4bf c[4]; uint4 q[2]; };
    Fr qf[4];
    {
        const __bf16* qrow = Qb + (size_t)(s0 + l16) * HIDDEN + h * HD;
        const int kb = half * 8;
        for (int dk = 0; dk < 4; ++dk) {
            qf[dk].q[0] = *(const uint4*)(qrow + dk * 32 + kb);
            qf[dk].q[1] = *(const uint4*)(qrow + dk * 32 + kb + 16);
        }
    }

    float mrow[8], lrow[8];
    v8f acc[8] = {};
    for (int r = 0; r < 8; ++r) { mrow[r] = -3.0e38f; lrow[r] = 0.f; }

    const float scale = 0.08838834764831845f;   // 1/sqrt(128)
    const __bf16* Kh = Kb + (size_t)kvh * SV * HD;
    const __bf16* Vh = Vb + (size_t)kvh * SV * HD;

    const int st = (tid * 4) >> 7;   // staging row 0..7 base
    const int sd = (tid * 4) & 127;  // staging col

    for (int t0 = 0; t0 < Lv; t0 += 32) {
        // cooperative stage: K tile [32][128] async->LDS, V^T tile [128][32] manual
        for (int it = 0; it < 4; ++it) {
            int t = st + it * 8;
            int trow = t0 + t; if (trow > SV - 1) trow = SV - 1;   // clamp, masked later
            async_copy_b64(&Kl[t][sd], Kh + (size_t)trow * HD + sd);
            v4bf vv = *(const v4bf*)(Vh + (size_t)trow * HD + sd);
            Vt[sd + 0][t] = vv[0]; Vt[sd + 1][t] = vv[1];
            Vt[sd + 2][t] = vv[2]; Vt[sd + 3][t] = vv[3];
        }
        if (t0 + 32 < Lv) {
            __builtin_prefetch(Kh + (size_t)(t0 + 32 + st) * HD + sd, 0, 1);
            __builtin_prefetch(Vh + (size_t)(t0 + 32 + st) * HD + sd, 0, 1);
        }
        async_wait0();
        __syncthreads();

        // S = Q @ K^T for this wave's 16 rows x 32 keys
        v8f sc[2];
        for (int nt = 0; nt < 2; ++nt) {
            v8f s = {};
            int trow = nt * 16 + l16;
            for (int dk = 0; dk < 4; ++dk) {
                Fr kf;
                const __bf16* kp = &Kl[trow][dk * 32 + half * 8];
                kf.c[0] = *(const v4bf*)(kp);
                kf.c[1] = *(const v4bf*)(kp + 4);
                kf.c[2] = *(const v4bf*)(kp + 16);
                kf.c[3] = *(const v4bf*)(kp + 20);
                s = __builtin_amdgcn_wmma_f32_16x16x32_bf16(
                        false, qf[dk].v, false, kf.v, (short)0, s, false, false);
            }
            sc[nt] = s;
        }

        const bool ok0 = (t0 + l16) < Lv;
        const bool ok1 = (t0 + 16 + l16) < Lv;
        float alpha[8];
        for (int r = 0; r < 8; ++r) {
            float a0 = ok0 ? sc[0][r] * scale : -3.0e38f;
            float a1 = ok1 ? sc[1][r] * scale : -3.0e38f;
            float mx = grp16_max(fmaxf(a0, a1));
            float mnew = fmaxf(mrow[r], mx);
            alpha[r] = __expf(mrow[r] - mnew);
            float p0 = __expf(a0 - mnew);
            float p1 = __expf(a1 - mnew);
            sc[0][r] = p0; sc[1][r] = p1;
            lrow[r] = lrow[r] * alpha[r] + grp16_sum(p0 + p1);
            mrow[r] = mnew;
        }
        for (int nd = 0; nd < 8; ++nd)
            for (int r = 0; r < 8; ++r) acc[nd][r] *= alpha[r];

        // route P through per-wave LDS: C-layout -> A-layout
        for (int r = 0; r < 8; ++r) {
            int row = half * 8 + r;
            Pl[w][row][l16]      = (__bf16)sc[0][r];
            Pl[w][row][16 + l16] = (__bf16)sc[1][r];
        }
        asm volatile("s_wait_dscnt 0" ::: "memory");   // same-wave DS RAW
        Fr pf;
        {
            const __bf16* pp = &Pl[w][l16][0];
            const int kb = half * 8;
            pf.c[0] = *(const v4bf*)(pp + kb);
            pf.c[1] = *(const v4bf*)(pp + kb + 4);
            pf.c[2] = *(const v4bf*)(pp + kb + 16);
            pf.c[3] = *(const v4bf*)(pp + kb + 20);
        }
        // O += P @ V
        for (int nd = 0; nd < 8; ++nd) {
            Fr vf;
            const __bf16* vp = &Vt[nd * 16 + l16][half * 8];
            vf.c[0] = *(const v4bf*)(vp);
            vf.c[1] = *(const v4bf*)(vp + 4);
            vf.c[2] = *(const v4bf*)(vp + 16);
            vf.c[3] = *(const v4bf*)(vp + 20);
            acc[nd] = __builtin_amdgcn_wmma_f32_16x16x32_bf16(
                          false, pf.v, false, vf.v, (short)0, acc[nd], false, false);
        }
        __syncthreads();
    }

    for (int r = 0; r < 8; ++r) {
        int srow = s0 + half * 8 + r;
        float f = ((srow >= start) && (hasimg != 0.f)) ? (1.f / lrow[r]) : 0.f;
        for (int nd = 0; nd < 8; ++nd)
            Ctx[(size_t)srow * HIDDEN + h * HD + nd * 16 + l16] = (__bf16)(acc[nd][r] * f);
    }
}

__global__ void fullrow_kernel(float* __restrict__ out,
                               const int* __restrict__ startp,
                               const float* __restrict__ hasimgp) {
    int s = blockIdx.x * blockDim.x + threadIdx.x;
    if (s < SLEN)
        out[s] = ((s >= startp[0]) && (hasimgp[0] != 0.f)) ? 1.f : 0.f;
}

// ---------------------------------------------------------------- launch
extern "C" void kernel_launch(void* const* d_in, const int* in_sizes, int n_in,
                              void* d_out, int out_size, void* d_ws, size_t ws_size,
                              hipStream_t stream) {
    const float* x      = (const float*)d_in[0];
    const float* vtok   = (const float*)d_in[1];
    const int*   vstart = (const int*)d_in[2];
    const int*   nchp   = (const int*)d_in[3];
    const float* hasimg = (const float*)d_in[4];
    const float* wq     = (const float*)d_in[5];
    const float* wk     = (const float*)d_in[6];
    const float* wv     = (const float*)d_in[7];
    const float* wo     = (const float*)d_in[8];
    const float* qnw    = (const float*)d_in[9];
    const float* knw    = (const float*)d_in[10];

    float* out_p  = (float*)d_out;                       // 1024*4096
    float* fr_p   = out_p + (size_t)SLEN * HIDDEN;       // 1024
    float* xk_p   = fr_p + SLEN;                         // 8*6404*128
    float* xv_p   = xk_p + (size_t)NKV * SV * HD;

    char* wsp = (char*)d_ws;
    auto alloc = [&](size_t bytes) {
        char* p = wsp;
        wsp += (bytes + 255) & ~(size_t)255;
        return p;
    };
    __bf16* x_bf  = (__bf16*)alloc((size_t)SLEN * HIDDEN * 2);
    __bf16* vt_bf = (__bf16*)alloc((size_t)SV * HIDDEN * 2);
    __bf16* wq_bf = (__bf16*)alloc((size_t)HIDDEN * HIDDEN * 2);
    __bf16* wk_bf = (__bf16*)alloc((size_t)HIDDEN * (NKV * HD) * 2);
    __bf16* wv_bf = (__bf16*)alloc((size_t)HIDDEN * (NKV * HD) * 2);
    __bf16* wo_bf = (__bf16*)alloc((size_t)HIDDEN * HIDDEN * 2);
    __bf16* xq_bf = (__bf16*)alloc((size_t)SLEN * HIDDEN * 2);
    __bf16* xk_bf = (__bf16*)alloc((size_t)NKV * SV * HD * 2);
    __bf16* xv_bf = (__bf16*)alloc((size_t)NKV * SV * HD * 2);
    __bf16* ctx_bf= (__bf16*)alloc((size_t)SLEN * HIDDEN * 2);

    auto cvt = [&](const float* s, __bf16* d, long n) {
        long n4 = n >> 2;
        int blocks = (int)((n4 + 255) / 256); if (blocks > 4096) blocks = 4096;
        cvt4_kernel<<<blocks, 256, 0, stream>>>((const float4*)s, (v4bf*)d, n4);
    };
    cvt(x,    x_bf,  (long)SLEN * HIDDEN);
    cvt(vtok, vt_bf, (long)SV * HIDDEN);
    cvt(wq,   wq_bf, (long)HIDDEN * HIDDEN);
    cvt(wk,   wk_bf, (long)HIDDEN * NKV * HD);
    cvt(wv,   wv_bf, (long)HIDDEN * NKV * HD);
    cvt(wo,   wo_bf, (long)HIDDEN * HIDDEN);

    // xq = rmsnorm(x @ wq) -> bf16 row-major
    gemm_bf16_kernel<<<dim3(SLEN / 128, HIDDEN / 128), 256, 0, stream>>>(
        x_bf, wq_bf, nullptr, xq_bf, qnw, HIDDEN, HIDDEN, SLEN, 1, 0);
    // xk = rmsnorm(vt @ wk) -> f32 output (head-major) + bf16 staging
    gemm_bf16_kernel<<<dim3((SV + 127) / 128, (NKV * HD) / 128), 256, 0, stream>>>(
        vt_bf, wk_bf, xk_p, xk_bf, knw, NKV * HD, HIDDEN, SV, 1, 1);
    // xv = vt @ wv -> f32 output (head-major) + bf16 staging
    gemm_bf16_kernel<<<dim3((SV + 127) / 128, (NKV * HD) / 128), 256, 0, stream>>>(
        vt_bf, wv_bf, xv_p, xv_bf, knw, NKV * HD, HIDDEN, SV, 0, 1);
    // flash attention -> ctx bf16 (rows pre-multiplied by full_row)
    xattn_kernel<<<dim3(NHEADS, SLEN / 128), 256, 0, stream>>>(
        xq_bf, xk_bf, xv_bf, ctx_bf, vstart, nchp, hasimg);
    // full_row output
    fullrow_kernel<<<dim3(SLEN / 256), 256, 0, stream>>>(fr_p, vstart, hasimg);
    // out = ctx @ wo -> f32
    gemm_bf16_kernel<<<dim3(SLEN / 128, HIDDEN / 128), 256, 0, stream>>>(
        ctx_bf, wo_bf, out_p, nullptr, qnw, HIDDEN, HIDDEN, SLEN, 0, 0);
}
